// GINGenerate_40802189312350
// MI455X (gfx1250) — matrix operand order repaired and checked by reading, not compiled, not run on hardware
//
#include <hip/hip_runtime.h>

typedef _Float16 v8h  __attribute__((ext_vector_type(8)));
typedef _Float16 v16h __attribute__((ext_vector_type(16)));
typedef float    v8f  __attribute__((ext_vector_type(8)));

#define DIM   300   // embedding dim
#define DPAD  320   // padded K / ld for node features (multiple of 32)
#define HID   600   // hidden dim (2*D)
#define HPAD  608   // padded hidden (19*32)
#define OPAD  304   // padded output cols of GEMM2 (19*16)
#define HID1  150   // lin1 out
#define OUTC  128   // final out

// ---------------- graph phase ----------------

// agg[n][d] = x[n][d] + ee1[4][d] + ee2[0][d]  (d<300), 0 for pads
__global__ void init_agg_kernel(const float* __restrict__ xin, int ldx,
                                const float* __restrict__ ee1l,
                                const float* __restrict__ ee2l,
                                float* __restrict__ agg, int N) {
  int n = blockIdx.x, d = threadIdx.x;
  if (n >= N) return;
  float v = 0.0f;
  if (d < DIM) v = xin[(size_t)n * ldx + d] + ee1l[4 * DIM + d] + ee2l[d];
  agg[(size_t)n * DPAD + d] = v;
}

// per-edge scatter-add: agg[dst] += x[src] + ee1[t0] + ee2[t1]
__global__ void scatter_kernel(const float* __restrict__ xin, int ldx,
                               const int* __restrict__ ei, const int* __restrict__ ea,
                               const float* __restrict__ ee1l,
                               const float* __restrict__ ee2l,
                               float* __restrict__ agg, int E) {
  int e = blockIdx.x, d = threadIdx.x;
  if (e >= E || d >= DIM) return;
  int src = ei[e];
  int dst = ei[E + e];
  int t0  = ea[2 * e];
  int t1  = ea[2 * e + 1];
  float v = xin[(size_t)src * ldx + d] + ee1l[t0 * DIM + d] + ee2l[t1 * DIM + d];
  atomicAdd(&agg[(size_t)dst * DPAD + d], v);
}

__global__ void cvt_f32_f16_kernel(const float* __restrict__ in,
                                   _Float16* __restrict__ out, int N) {
  int n = blockIdx.x, d = threadIdx.x;
  if (n >= N) return;
  out[(size_t)n * DPAD + d] = (_Float16)in[(size_t)n * DPAD + d];
}

// W:[K][O] row-major -> WT:[Opad][Kpad] f16, zero padded (pads inert in GEMM)
__global__ void conv_wT_kernel(const float* __restrict__ W, _Float16* __restrict__ WT,
                               int K, int O, int Kpad) {
  int o = blockIdx.x, k = threadIdx.x;
  _Float16 v = (_Float16)0.0f;
  if (o < O && k < K) v = (_Float16)W[(size_t)k * O + o];
  WT[(size_t)o * Kpad + k] = v;
}

// ---------------- WMMA GEMM ----------------
// One wave computes a 16-row x 64-col tile (4 accumulators, A reused).
// A: f16 [M][lda], row-major. BT: f16 [Ncols][ldb] (B transposed), row-major.
// A frag layout (16x32 f16): lanes 0-15: elems 0..7 = K+0..7, 8..15 = K+16..23;
//                            lanes 16-31: +8 on each K base.
// B frag layout (32x16 f16): same K mapping as A (lanes 0-15: K+0..7, K+16..23; lanes 16-31: +8).
template <bool RELU, bool OUTH>
__global__ __launch_bounds__(32) void wmma_gemm_kernel(
    const _Float16* __restrict__ A, int lda,
    const _Float16* __restrict__ BT, int ldb,
    const float* __restrict__ bias, int bias_n,
    void* __restrict__ Cout, int ldc,
    int M, int Kpad, int ntilesN) {
  const int  lane  = threadIdx.x;
  const int  l15   = lane & 15;
  const bool hi    = (lane & 16) != 0;
  const int  mtile = blockIdx.x;
  const int  tc0   = blockIdx.y * 4;

  int mrow  = mtile * 16 + l15;
  int mload = mrow < M ? mrow : (M - 1);
  const int koffA = hi ? 8 : 0;
  const int koffB = hi ? 8 : 0;
  const int ncmax = ntilesN * 16;

  int nc[4];
#pragma unroll
  for (int j = 0; j < 4; ++j) {
    int c = (tc0 + j) * 16 + l15;
    nc[j] = (c < ncmax) ? c : l15;  // clamp invalid tiles to safe addresses
  }

  v8f acc[4] = {};

  for (int k = 0; k < Kpad; k += 32) {
    const _Float16* ap = A + (size_t)mload * lda + (k + koffA);
    v8h a0 = *(const v8h*)(ap);
    v8h a1 = *(const v8h*)(ap + 16);
    v16h a;
#pragma unroll
    for (int i = 0; i < 8; ++i) { a[i] = a0[i]; a[i + 8] = a1[i]; }

#pragma unroll
    for (int j = 0; j < 4; ++j) {
      const _Float16* bp = BT + (size_t)nc[j] * ldb + (k + koffB);
      v8h b0 = *(const v8h*)(bp);
      v8h b1 = *(const v8h*)(bp + 16);
      v16h b;
#pragma unroll
      for (int i = 0; i < 8; ++i) { b[i] = b0[i]; b[i + 8] = b1[i]; }
      acc[j] = __builtin_amdgcn_wmma_f32_16x16x32_f16(
          false, a, false, b, (short)0, acc[j], false, false);
    }
  }

  // epilogue: bias (+ReLU), scatter C fragment (VGPR i -> row i or i+8)
#pragma unroll
  for (int j = 0; j < 4; ++j) {
    int ct = tc0 + j;
    if (ct >= ntilesN) continue;
    int col = ct * 16 + l15;
    float bv = (col < bias_n) ? bias[col] : 0.0f;
#pragma unroll
    for (int i = 0; i < 8; ++i) {
      int row = mtile * 16 + i + (hi ? 8 : 0);
      if (row >= M) continue;
      float v = acc[j][i] + bv;
      if (RELU) v = v > 0.0f ? v : 0.0f;
      if (OUTH)
        ((_Float16*)Cout)[(size_t)row * ldc + col] = (_Float16)v;
      else
        ((float*)Cout)[(size_t)row * ldc + col] = v;
    }
  }
}

// ---------------- pooling + head ----------------

__global__ void zero_g_kernel(float* __restrict__ g, int G) {
  int r = blockIdx.x, d = threadIdx.x;
  if (r < G && d < DIM) g[(size_t)r * DIM + d] = 0.0f;
}

__global__ void pool_kernel(const float* __restrict__ h, int ldh,
                            const int* __restrict__ batch,
                            float* __restrict__ g, int N) {
  int n = blockIdx.x, d = threadIdx.x;
  if (n >= N || d >= DIM) return;
  atomicAdd(&g[(size_t)batch[n] * DIM + d], h[(size_t)n * ldh + d]);
}

__global__ void head1_kernel(const float* __restrict__ g,
                             const float* __restrict__ W,   // [300][150]
                             const float* __restrict__ b,   // [150]
                             float* __restrict__ g2) {
  int r = blockIdx.x, c = threadIdx.x;
  if (c >= HID1) return;
  float acc = b[c];
  const float* gr = g + (size_t)r * DIM;
  for (int k = 0; k < DIM; ++k) acc += gr[k] * W[(size_t)k * HID1 + c];
  g2[(size_t)r * HID1 + c] = acc > 0.0f ? acc : 0.0f;
}

__global__ void head2_kernel(const float* __restrict__ g2,
                             const float* __restrict__ W,   // [150][128]
                             const float* __restrict__ b,   // [128]
                             float* __restrict__ out) {
  int r = blockIdx.x, c = threadIdx.x;
  if (c >= OUTC) return;
  float acc = b[c];
  const float* gr = g2 + (size_t)r * HID1;
  for (int k = 0; k < HID1; ++k) acc += gr[k] * W[(size_t)k * OUTC + c];
  out[(size_t)r * OUTC + c] = acc;
}

// ---------------- host ----------------

extern "C" void kernel_launch(void* const* d_in, const int* in_sizes, int n_in,
                              void* d_out, int out_size, void* d_ws, size_t ws_size,
                              hipStream_t stream) {
  const float* x   = (const float*)d_in[0];
  const float* W1  = (const float*)d_in[1];
  const float* b1  = (const float*)d_in[2];
  const float* W2  = (const float*)d_in[3];
  const float* b2  = (const float*)d_in[4];
  const float* ee1 = (const float*)d_in[5];
  const float* ee2 = (const float*)d_in[6];
  const float* l1W = (const float*)d_in[7];
  const float* l1b = (const float*)d_in[8];
  const float* l2W = (const float*)d_in[9];
  const float* l2b = (const float*)d_in[10];
  const int* ei    = (const int*)d_in[11];
  const int* ea    = (const int*)d_in[12];
  const int* batch = (const int*)d_in[13];

  const int N = in_sizes[13];
  const int E = in_sizes[11] / 2;
  const int G = out_size / OUTC;

  // carve workspace
  char*  w   = (char*)d_ws;
  size_t off = 0;
  auto carve = [&](size_t bytes) -> void* {
    void* p = w + off;
    off = (off + bytes + 255) & ~(size_t)255;
    return p;
  };
  float*    agg  = (float*)carve((size_t)N * DPAD * sizeof(float));
  _Float16* aggh = (_Float16*)carve((size_t)N * DPAD * sizeof(_Float16));
  _Float16* tmp  = (_Float16*)carve((size_t)N * HPAD * sizeof(_Float16));
  float*    hA   = (float*)carve((size_t)N * DPAD * sizeof(float));
  float*    hB   = (float*)carve((size_t)N * DPAD * sizeof(float));
  _Float16* W1T  = (_Float16*)carve((size_t)5 * HPAD * DPAD * sizeof(_Float16));
  _Float16* W2T  = (_Float16*)carve((size_t)5 * OPAD * HPAD * sizeof(_Float16));
  float*    gbuf = (float*)carve((size_t)G * DIM * sizeof(float));
  float*    g2b  = (float*)carve((size_t)G * HID1 * sizeof(float));

  const int mtiles = (N + 15) / 16;
  const int nt1 = HPAD / 16;  // 38 column tiles, GEMM1
  const int nt2 = OPAD / 16;  // 19 column tiles, GEMM2

  const float* hin = x;
  int ldin = DIM;

  for (int l = 0; l < 5; ++l) {
    const float* ee1l = ee1 + (size_t)l * 6 * DIM;
    const float* ee2l = ee2 + (size_t)l * 3 * DIM;
    _Float16* w1t = W1T + (size_t)l * HPAD * DPAD;
    _Float16* w2t = W2T + (size_t)l * OPAD * HPAD;

    conv_wT_kernel<<<HPAD, DPAD, 0, stream>>>(W1 + (size_t)l * DIM * HID, w1t, DIM, HID, DPAD);
    conv_wT_kernel<<<OPAD, HPAD, 0, stream>>>(W2 + (size_t)l * HID * DIM, w2t, HID, DIM, HPAD);

    init_agg_kernel<<<N, DPAD, 0, stream>>>(hin, ldin, ee1l, ee2l, agg, N);
    scatter_kernel<<<E, DPAD, 0, stream>>>(hin, ldin, ei, ea, ee1l, ee2l, agg, E);
    cvt_f32_f16_kernel<<<N, DPAD, 0, stream>>>(agg, aggh, N);

    float* hout = (l & 1) ? hB : hA;
    dim3 grid1(mtiles, (nt1 + 3) / 4);
    wmma_gemm_kernel<true, true><<<grid1, 32, 0, stream>>>(
        aggh, DPAD, w1t, DPAD, b1 + (size_t)l * HID, HID, tmp, HPAD, N, DPAD, nt1);
    dim3 grid2(mtiles, (nt2 + 3) / 4);
    wmma_gemm_kernel<false, false><<<grid2, 32, 0, stream>>>(
        tmp, HPAD, w2t, HPAD, b2 + (size_t)l * DIM, DIM, hout, DPAD, N, HPAD, nt2);

    hin = hout;
    ldin = DPAD;
  }

  zero_g_kernel<<<G, DPAD, 0, stream>>>(gbuf, G);
  pool_kernel<<<N, DPAD, 0, stream>>>(hin, ldin, batch, gbuf, N);
  head1_kernel<<<G, 160, 0, stream>>>(gbuf, l1W, l1b, g2b);
  head2_kernel<<<G, OUTC, 0, stream>>>(g2b, l2W, l2b, (float*)d_out);
}
